// Net_53515292508160
// MI455X (gfx1250) — matrix operand order repaired and checked by reading, not compiled, not run on hardware
//
#include <hip/hip_runtime.h>
#include <hip/hip_bf16.h>

typedef __attribute__((ext_vector_type(16))) __bf16 v16bf;
typedef __attribute__((ext_vector_type(8)))  __bf16 v8bf;
typedef __attribute__((ext_vector_type(8)))  float  v8f;

#define KSP   5
#define NKER  125   // 5^3

__device__ __forceinline__ float elu1(float v) {
    return v > 0.0f ? v : (expf(v) - 1.0f);
}

// ---------------------------------------------------------------- utilities
__global__ void fill_zero_u32(unsigned* __restrict__ p, long long n) {
    long long i = (long long)blockIdx.x * blockDim.x + threadIdx.x;
    if (i < n) p[i] = 0u;
}

__global__ void cvt_bf16(const float* __restrict__ in, __bf16* __restrict__ out,
                         long long n) {
    long long i = (long long)blockIdx.x * blockDim.x + threadIdx.x;
    if (i < n) out[i] = (__bf16)in[i];
}

// W [nk][Cin][Cout] f32  ->  Wt [nk][Cout][Cin] bf16   (B operand, K contiguous)
__global__ void wtrans(const float* __restrict__ W, __bf16* __restrict__ Wt,
                       int nk, int Cin, int Cout) {
    long long i = (long long)blockIdx.x * blockDim.x + threadIdx.x;
    long long total = (long long)nk * Cin * Cout;
    if (i >= total) return;
    int c   = (int)(i % Cin);
    long long r = i / Cin;
    int d   = (int)(r % Cout);
    int k   = (int)(r / Cout);
    Wt[i] = (__bf16)W[((long long)k * Cin + c) * Cout + d];
}

// ------------------------------------------------------- spline basis + deg
__global__ void basis_kernel(const float* __restrict__ pseudo,
                             const int* __restrict__ dst,
                             float* __restrict__ basis, int* __restrict__ widx,
                             float* __restrict__ deg, int E) {
    int e = blockIdx.x * blockDim.x + threadIdx.x;
    if (e >= E) return;
    float fr[3]; int bt[3];
    #pragma unroll
    for (int d = 0; d < 3; d++) {
        float v  = pseudo[e * 3 + d] * (float)(KSP - 1);
        float fl = floorf(v);
        fl = fminf(fmaxf(fl, 0.0f), (float)(KSP - 2));
        bt[d] = (int)fl;
        fr[d] = v - fl;
    }
    const int stride0 = 1, stride1 = KSP, stride2 = KSP * KSP;
    #pragma unroll
    for (int s = 0; s < 8; s++) {
        int b0 = s & 1, b1 = (s >> 1) & 1, b2 = (s >> 2) & 1;
        float w = (b0 ? fr[0] : 1.0f - fr[0]) *
                  (b1 ? fr[1] : 1.0f - fr[1]) *
                  (b2 ? fr[2] : 1.0f - fr[2]);
        int idx = (bt[0] + b0) * stride0 + (bt[1] + b1) * stride1 +
                  (bt[2] + b2) * stride2;
        basis[e * 8 + s] = w;
        widx [e * 8 + s] = idx;
    }
    atomicAdd(&deg[dst[e]], 1.0f);
}

// ------------------------------------------------------------- WMMA GEMM
// C(out) = A[M,K] @ Bt[k][Ncols,K]^T   for k = blockIdx.y
//   A  : bf16 row-major [M,K]
//   Bt : bf16 [nk][Ncols][K]  (output column -> contiguous K)
//   out element (k, m, d) at  kStride*k + m*ldC + d   (f32 or bf16)
__global__ void __launch_bounds__(256)
wmma_gemm(const __bf16* __restrict__ A, const __bf16* __restrict__ Bt,
          float* __restrict__ Cf, __bf16* __restrict__ Cb,
          int M, int Kdim, int Ncols, long long ldC, long long kStride,
          int bf16out) {
    int k = blockIdx.y;
    const __bf16* B = Bt + (size_t)k * Kdim * Ncols;

    int lane = threadIdx.x & 31;
    int wave = threadIdx.x >> 5;
    int tilesN = Ncols >> 4;
    int tilesM = (M + 15) >> 4;
    int tile = blockIdx.x * (blockDim.x >> 5) + wave;
    if (tile >= tilesM * tilesN) return;          // wave-uniform exit
    int mt = tile / tilesN, nt = tile % tilesN;

    int lh = lane & 15;       // row (A) / col (B,C) within tile
    int hi = lane >> 4;       // half select

    int m = mt * 16 + lh; if (m >= M) m = M - 1;  // clamp loads (rows guarded on store)
    int d = nt * 16 + lh;

    const __bf16* arow = A + (size_t)m * Kdim + hi * 16;  // A: contiguous K block of 16
    const __bf16* brow = B + (size_t)d * Kdim + hi * 16;  // B: contiguous K block of 16

    v8f acc = {};
    for (int k0 = 0; k0 < Kdim; k0 += 32) {
        v8bf alo = *(const v8bf*)(arow + k0);        // K = k0 + hi*16 .. +7
        v8bf ahi = *(const v8bf*)(arow + k0 + 8);    // K = k0 + hi*16 + 8 .. +15
        v16bf a, b;
        v16bf bq = *(const v16bf*)(brow + k0);       // K = k0 + hi*16 .. +15
        #pragma unroll
        for (int i = 0; i < 8; i++) { a[i] = alo[i]; a[i + 8] = ahi[i]; }
        #pragma unroll
        for (int i = 0; i < 16; i++) { b[i] = bq[i]; }
        acc = __builtin_amdgcn_wmma_f32_16x16x32_bf16(
                  false, a, false, b, (short)0, acc, false, false);
    }

    long long base = kStride * (long long)k + (long long)(nt * 16 + lh);
    #pragma unroll
    for (int r = 0; r < 8; r++) {
        int mr = mt * 16 + hi * 8 + r;
        if (mr < M) {
            long long off = base + (long long)mr * ldC;
            if (bf16out) Cb[off] = (__bf16)acc[r];
            else         Cf[off] = acc[r];
        }
    }
}

// ---------------------------------------------------- edge gather + scatter
__global__ void msg_scatter(const float* __restrict__ basis,
                            const int* __restrict__ widx,
                            const int* __restrict__ src,
                            const int* __restrict__ dst,
                            const __bf16* __restrict__ Y,
                            float* __restrict__ agg,
                            int E, int coutShift) {
    long long tid = (long long)blockIdx.x * blockDim.x + threadIdx.x;
    if (tid >= ((long long)E << coutShift)) return;
    int e = (int)(tid >> coutShift);
    int d = (int)(tid & ((1 << coutShift) - 1));
    int sn = src[e];
    int b0 = e * 8;
    float m = 0.0f;
    #pragma unroll
    for (int s = 0; s < 8; s++) {
        int g = sn * NKER + widx[b0 + s];
        m += basis[b0 + s] * (float)Y[((long long)g << coutShift) + d];
    }
    atomicAdd(&agg[((long long)dst[e] << coutShift) + d], m);
}

// ------------------------------------------------------------- node updates
__global__ void node_update(const float* __restrict__ agg,
                            const float* __restrict__ deg,
                            const float* __restrict__ R,
                            const float* __restrict__ bias,
                            float* __restrict__ hout,
                            long long total, int coutShift) {
    long long tid = (long long)blockIdx.x * blockDim.x + threadIdx.x;
    if (tid >= total) return;
    int d = (int)(tid & ((1 << coutShift) - 1));
    long long n = tid >> coutShift;
    float a = agg[tid] / fmaxf(deg[n], 1.0f);
    hout[tid] = elu1(a + R[tid] + bias[d]);
}

// ---- layer 1 (Cin = 1): outer product, no GEMM needed
__global__ void l1_msg(const float* __restrict__ x,
                       const int* __restrict__ src, const int* __restrict__ dst,
                       const float* __restrict__ basis,
                       const int* __restrict__ widx,
                       const float* __restrict__ W1, float* __restrict__ agg,
                       int E) {
    long long tid = (long long)blockIdx.x * blockDim.x + threadIdx.x;
    if (tid >= ((long long)E << 5)) return;
    int e = (int)(tid >> 5), d = (int)(tid & 31);
    float xs = x[src[e]];
    int b0 = e * 8;
    float s = 0.0f;
    #pragma unroll
    for (int k = 0; k < 8; k++)
        s += basis[b0 + k] * W1[widx[b0 + k] * 32 + d];
    atomicAdd(&agg[((long long)dst[e] << 5) + d], xs * s);
}

__global__ void l1_update(const float* __restrict__ agg,
                          const float* __restrict__ deg,
                          const float* __restrict__ x,
                          const float* __restrict__ root1,
                          const float* __restrict__ b1,
                          float* __restrict__ hout, int N) {
    long long tid = (long long)blockIdx.x * blockDim.x + threadIdx.x;
    if (tid >= ((long long)N << 5)) return;
    int d = (int)(tid & 31);
    long long n = tid >> 5;
    float a = agg[tid] / fmaxf(deg[n], 1.0f);
    hout[tid] = elu1(a + x[n] * root1[d] + b1[d]);
}

// ---- MLP epilogues
__global__ void bias_elu_bf16(const float* __restrict__ in,
                              const float* __restrict__ bias,
                              __bf16* __restrict__ out,
                              long long total, int mask) {
    long long tid = (long long)blockIdx.x * blockDim.x + threadIdx.x;
    if (tid >= total) return;
    out[tid] = (__bf16)elu1(in[tid] + bias[(int)(tid & mask)]);
}

__global__ void final_logsoftmax(const float* __restrict__ t,
                                 const float* __restrict__ lb2,
                                 float* __restrict__ out, int N) {
    int n = blockIdx.x * blockDim.x + threadIdx.x;
    if (n >= N) return;
    float v[4];
    #pragma unroll
    for (int d = 0; d < 4; d++) v[d] = t[(long long)n * 16 + d] + lb2[d];
    float m = fmaxf(fmaxf(v[0], v[1]), fmaxf(v[2], v[3]));
    float s = 0.0f;
    #pragma unroll
    for (int d = 0; d < 4; d++) s += expf(v[d] - m);
    float l = logf(s);
    #pragma unroll
    for (int d = 0; d < 4; d++) out[(long long)n * 4 + d] = v[d] - m - l;
}

// =========================================================================
extern "C" void kernel_launch(void* const* d_in, const int* in_sizes, int n_in,
                              void* d_out, int out_size, void* d_ws, size_t ws_size,
                              hipStream_t stream) {
    (void)n_in; (void)out_size; (void)ws_size;
    const float* x      = (const float*)d_in[0];
    const float* pseudo = (const float*)d_in[1];
    const int*   eidx   = (const int*)d_in[2];
    const float* W1     = (const float*)d_in[3];
    const float* root1  = (const float*)d_in[4];
    const float* b1     = (const float*)d_in[5];
    const float* W2     = (const float*)d_in[6];
    const float* root2  = (const float*)d_in[7];
    const float* b2     = (const float*)d_in[8];
    const float* Wl     = (const float*)d_in[9];
    const float* rootl  = (const float*)d_in[10];
    const float* bl     = (const float*)d_in[11];
    const float* lw1    = (const float*)d_in[12];
    const float* lb1    = (const float*)d_in[13];
    const float* lw2    = (const float*)d_in[14];
    const float* lb2    = (const float*)d_in[15];

    const int N = in_sizes[0];          // x is [N,1]
    const int E = in_sizes[1] / 3;      // pseudo [E,3]
    const int* srcI = eidx;
    const int* dstI = eidx + E;

    // ---- workspace layout (256B aligned slices)
    char* ws = (char*)d_ws;
    size_t off = 0;
    auto salloc = [&](size_t bytes) -> void* {
        void* p = ws + off;
        off = (off + bytes + 255) & ~(size_t)255;
        return p;
    };
    float*  basis = (float*) salloc((size_t)E * 8 * 4);
    int*    widx  = (int*)   salloc((size_t)E * 8 * 4);
    float*  deg   = (float*) salloc((size_t)N * 4);
    __bf16* Y     = (__bf16*)salloc((size_t)N * NKER * 64 * 2);   // 160 MB, ~L2 sized
    __bf16* Hb    = (__bf16*)salloc((size_t)N * 128 * 2);
    __bf16* Wt    = (__bf16*)salloc((size_t)NKER * 64 * 64 * 2);
    __bf16* rootT = (__bf16*)salloc((size_t)64 * 64 * 2);
    float*  agg   = (float*) salloc((size_t)N * 64 * 4);          // also MLP2 out [N,16]
    float*  tmpR  = (float*) salloc((size_t)N * 128 * 4);         // root term / MLP1 out
    float*  hA    = (float*) salloc((size_t)N * 64 * 4);
    float*  hB    = (float*) salloc((size_t)N * 64 * 4);

    auto nblk = [](long long n) { return (unsigned)((n + 255) / 256); };
    auto zero = [&](void* p, long long words) {
        fill_zero_u32<<<nblk(words), 256, 0, stream>>>((unsigned*)p, words);
    };

    // ---- basis + degree (shared by all layers)
    zero(deg, N);
    basis_kernel<<<nblk(E), 256, 0, stream>>>(pseudo, dstI, basis, widx, deg, E);

    // ---- layer 1: Cin=1, Cout=32 (outer product path)
    zero(agg, (long long)N * 32);
    l1_msg<<<nblk((long long)E * 32), 256, 0, stream>>>(x, srcI, dstI, basis,
                                                        widx, W1, agg, E);
    l1_update<<<nblk((long long)N * 32), 256, 0, stream>>>(agg, deg, x, root1,
                                                           b1, hA, N);

    const int tilesM = (N + 15) / 16;

    // ---- generic WMMA conv layer
    auto conv = [&](const float* hin, float* hout, const float* W,
                    const float* root, const float* bias, int Cin, int Cout,
                    int coutShift) {
        long long nHin = (long long)N * Cin;
        cvt_bf16<<<nblk(nHin), 256, 0, stream>>>(hin, Hb, nHin);
        long long nW = (long long)NKER * Cin * Cout;
        wtrans<<<nblk(nW), 256, 0, stream>>>(W, Wt, NKER, Cin, Cout);
        wtrans<<<nblk((long long)Cin * Cout), 256, 0, stream>>>(root, rootT, 1,
                                                                Cin, Cout);
        zero(agg, (long long)N * Cout);
        int tiles = tilesM * (Cout / 16);
        dim3 gY((tiles + 7) / 8, NKER);
        wmma_gemm<<<gY, 256, 0, stream>>>(Hb, Wt, nullptr, Y, N, Cin, Cout,
                                          (long long)NKER * Cout,
                                          (long long)Cout, 1);
        dim3 gR((tiles + 7) / 8, 1);
        wmma_gemm<<<gR, 256, 0, stream>>>(Hb, rootT, tmpR, nullptr, N, Cin,
                                          Cout, (long long)Cout, 0, 0);
        msg_scatter<<<nblk((long long)E << coutShift), 256, 0, stream>>>(
            basis, widx, srcI, dstI, Y, agg, E, coutShift);
        node_update<<<nblk((long long)N << coutShift), 256, 0, stream>>>(
            agg, deg, tmpR, bias, hout, (long long)N << coutShift, coutShift);
    };

    // ---- layer 2: 32 -> 64
    conv(hA, hB, W2, root2, b2, 32, 64, 6);

    // ---- layers 3..10: 64 -> 64
    float* cur = hB;
    float* nxt = hA;
    for (int i = 0; i < 8; i++) {
        conv(cur, nxt, Wl + (size_t)i * NKER * 64 * 64,
             rootl + (size_t)i * 64 * 64, bl + (size_t)i * 64, 64, 64, 6);
        float* t = cur; cur = nxt; nxt = t;
    }

    // ---- MLP1: [N,64] @ [64,128] + b, ELU  (WMMA)
    cvt_bf16<<<nblk((long long)N * 64), 256, 0, stream>>>(cur, Hb,
                                                          (long long)N * 64);
    wtrans<<<nblk(64 * 128), 256, 0, stream>>>(lw1, Wt, 1, 64, 128);
    {
        int tiles = tilesM * 8;
        dim3 g((tiles + 7) / 8, 1);
        wmma_gemm<<<g, 256, 0, stream>>>(Hb, Wt, tmpR, nullptr, N, 64, 128,
                                         128, 0, 0);
    }
    bias_elu_bf16<<<nblk((long long)N * 128), 256, 0, stream>>>(
        tmpR, lb1, Hb, (long long)N * 128, 127);

    // ---- MLP2: [N,128] @ [128,4] padded to 16 cols (WMMA)
    zero(Wt, 16 * 128 * 2 / 4);                        // zero-pad rows 4..15
    wtrans<<<nblk(128 * 4), 256, 0, stream>>>(lw2, Wt, 1, 128, 4);
    {
        int tiles = tilesM * 1;
        dim3 g((tiles + 7) / 8, 1);
        wmma_gemm<<<g, 256, 0, stream>>>(Hb, Wt, agg, nullptr, N, 128, 16, 16,
                                         0, 0);
    }
    final_logsoftmax<<<nblk(N), 256, 0, stream>>>(agg, lb2, (float*)d_out, N);
}
